// SelfAttention_37744172597494
// MI455X (gfx1250) — matrix operand order repaired and checked by reading, not compiled, not run on hardware
//
#include <hip/hip_runtime.h>
#include <hip/hip_bf16.h>

// ---------------------------------------------------------------------------
// Self-attention for B=8, S=2048, D=768 on gfx1250 (MI455X).
// bf16 WMMA (v_wmma_f32_16x16x32_bf16) for all GEMMs, f32 softmax, flash-style
// online softmax so scores are never materialized in global memory.
// ---------------------------------------------------------------------------

typedef __attribute__((ext_vector_type(16))) __bf16 v16bf;
typedef __attribute__((ext_vector_type(8)))  float  v8f;

#define SEQ   2048
#define DIM   768
#define NROW  16384           // B*S
#define NRT   1024            // row tiles of 16
#define NDS   24              // DIM / 32 (WMMA K-steps)
#define NET   48              // DIM / 16 (output col tiles)
#define FRAGI 256             // dwords per fragment (32 lanes x 8 dwords)

union FragBF { int i[8]; int4 q[2]; v16bf v; };
union FragF  { float f[8]; v8f v; };

static __device__ inline v8f vzero8() { v8f z = {0.f,0.f,0.f,0.f,0.f,0.f,0.f,0.f}; return z; }

static __device__ inline v8f wmma_bf16(const FragBF& a, const FragBF& b, v8f c) {
  return __builtin_amdgcn_wmma_f32_16x16x32_bf16(false, a.v, false, b.v,
                                                 (short)0, c, false, false);
}

// RNE float -> bf16 bit pattern
static __device__ inline unsigned int f2bf(float f) {
  unsigned int u = __float_as_uint(f);
  return ((u + 0x7FFFu + ((u >> 16) & 1u)) >> 16) & 0xFFFFu;
}
static __device__ inline unsigned int pack2bf(float a, float b) {
  return f2bf(a) | (f2bf(b) << 16);
}

// A-matrix (16x32 bf16) K index for (lane-half hi, vgpr r): even base of pair
static __device__ inline int kmapA(int hi, int r) {
  return 16 * hi + 2 * r;
}
// B-matrix (32x16 bf16) K index: lanes 0-15 hold K=0..15, lanes 16-31 K=16..31
static __device__ inline int kmapB(int hi, int r) { return 16 * hi + 2 * r; }

// ---------------------------------------------------------------------------
// Kernel 1: convert Wq/Wk/Wv (f32 row-major DxD) into raw B-fragment tiles.
// wfrag dword index: ((m*48 + e)*24 + j)*256 + lane*8 + r
// ---------------------------------------------------------------------------
__global__ __launch_bounds__(32) void wfrag_kernel(const float* __restrict__ Wq,
                                                   const float* __restrict__ Wk,
                                                   const float* __restrict__ Wv,
                                                   int* __restrict__ wfrag) {
  const int fid = blockIdx.x;                 // 0 .. 3*48*24-1
  const int m   = fid / (NET * NDS);
  const int rem = fid % (NET * NDS);
  const int e   = rem / NDS;
  const int j   = rem % NDS;
  const float* W = (m == 0) ? Wq : ((m == 1) ? Wk : Wv);
  const int lane = threadIdx.x & 31;
  const int n = lane & 15, hi = lane >> 4;
  FragBF fr;
#pragma unroll
  for (int r = 0; r < 8; ++r) {
    const int kf = j * 32 + kmapB(hi, r);
    const float f0 = W[kf * DIM + e * 16 + n];
    const float f1 = W[(kf + 1) * DIM + e * 16 + n];
    fr.i[r] = pack2bf(f0, f1);
  }
  int* dst = wfrag + (fid << 8) + lane * 8;
  *(int4*)dst       = fr.q[0];
  *((int4*)dst + 1) = fr.q[1];
}

// ---------------------------------------------------------------------------
// Kernel 2: QKV projection. One workgroup (8 waves) per 16-row tile.
//   Q -> raw A-fragments   qfrag[(rt*24 + j)*256 + lane*8 + r]
//   K -> raw B-fragments   kfrag[(rt*24 + j)*256 + lane*8 + r]
//   V -> seq-pair-interleaved bf16: Vp dword[sp*768 + d] = {V[2sp,d], V[2sp+1,d]}
// ---------------------------------------------------------------------------
__global__ __launch_bounds__(256) void qkv_proj_kernel(
    const float* __restrict__ x,  const float* __restrict__ bq,
    const float* __restrict__ bk, const float* __restrict__ bv,
    const int*  __restrict__ wfrag,
    int* __restrict__ qfrag, int* __restrict__ kfrag,
    unsigned int* __restrict__ Vp) {
  __shared__ __align__(16) unsigned short xs[16 * DIM];     // 24 KB bf16 x-tile
  __shared__ __align__(16) unsigned short stg[8][16 * 32];  // per-wave staging

  const int tid = threadIdx.x, lane = tid & 31, wave = tid >> 5;
  const int n = lane & 15, hi = lane >> 4;
  const int rt = blockIdx.x;

  for (int i = tid; i < 16 * DIM; i += 256)
    xs[i] = (unsigned short)f2bf(x[rt * (16 * DIM) + i]);
  __syncthreads();

  // 96 jobs: 24 Q-fragment jobs, 24 K-fragment jobs, 48 V-tile jobs
  for (int job = wave; job < 96; job += 8) {
    if (job < 48) {
      const int isK = (job >= 24) ? 1 : 0;
      const int j = isK ? (job - 24) : job;     // feature 32-chunk index
      const float* bias = isK ? bk : bq;
      const int* wbase = wfrag + isK * (NET * NDS * FRAGI);  // m=0 or m=1
      for (int half = 0; half < 2; ++half) {
        const int e = j * 2 + half;
        FragF acc; acc.v = vzero8();
        for (int ds = 0; ds < NDS; ++ds) {
          FragBF a, bf;
          const int abase = n * DIM + ds * 32;
#pragma unroll
          for (int r = 0; r < 8; ++r)
            a.i[r] = *(const int*)&xs[abase + kmapA(hi, r)];
          const int4* wp = (const int4*)(wbase + ((e * NDS + ds) << 8) + lane * 8);
          bf.q[0] = wp[0]; bf.q[1] = wp[1];
          acc.v = wmma_bf16(a, bf, acc.v);
        }
        const float bb = bias[e * 16 + n];
#pragma unroll
        for (int r = 0; r < 8; ++r)
          stg[wave][(r + 8 * hi) * 32 + half * 16 + n] =
              (unsigned short)f2bf(acc.f[r] + bb);
      }
      asm volatile("s_wait_dscnt 0" ::: "memory");  // wave-local LDS transpose fence
      FragBF fr;
      if (!isK) {   // Q: A-fragment (M = seq row, K = feature)
#pragma unroll
        for (int r = 0; r < 8; ++r)
          fr.i[r] = *(const int*)&stg[wave][n * 32 + kmapA(hi, r)];
      } else {      // K: B-fragment (N = seq col, K = feature)
#pragma unroll
        for (int r = 0; r < 8; ++r)
          fr.i[r] = *(const int*)&stg[wave][n * 32 + kmapB(hi, r)];
      }
      int* dst = (isK ? kfrag : qfrag) + ((rt * NDS + j) << 8) + lane * 8;
      *(int4*)dst       = fr.q[0];
      *((int4*)dst + 1) = fr.q[1];
      asm volatile("s_wait_dscnt 0" ::: "memory");
    } else {
      // V tile, output cols e*16..e*16+15
      const int e = job - 48;
      const int* wbase = wfrag + 2 * (NET * NDS * FRAGI);
      FragF acc; acc.v = vzero8();
      for (int ds = 0; ds < NDS; ++ds) {
        FragBF a, bf;
        const int abase = n * DIM + ds * 32;
#pragma unroll
        for (int r = 0; r < 8; ++r)
          a.i[r] = *(const int*)&xs[abase + kmapA(hi, r)];
        const int4* wp = (const int4*)(wbase + ((e * NDS + ds) << 8) + lane * 8);
        bf.q[0] = wp[0]; bf.q[1] = wp[1];
        acc.v = wmma_bf16(a, bf, acc.v);
      }
      const float bb = bv[e * 16 + n];
      unsigned short* VpH = (unsigned short*)Vp;
#pragma unroll
      for (int r = 0; r < 8; ++r) {
        const int s = rt * 16 + r + 8 * hi;             // global sequence row
        VpH[((unsigned)(s >> 1) * DIM + e * 16 + n) * 2 + (s & 1)] =
            (unsigned short)f2bf(acc.f[r] + bb);
      }
    }
  }
}

// ---------------------------------------------------------------------------
// Kernel 3: flash attention + residual. One workgroup per 16-row q-tile.
// k-tiles of 128 cols: 8 waves each compute one 16x16 score tile (24 WMMAs),
// ALL 256 threads cooperate on the online softmax (one row per 16 threads,
// shfl_xor reductions), then each wave updates its 96-col d-chunk of the
// output accumulator with 24 P*V WMMAs.
// ---------------------------------------------------------------------------
__global__ __launch_bounds__(256) void attn_kernel(
    const float* __restrict__ x, const int* __restrict__ qfrag,
    const int* __restrict__ kfrag, const unsigned int* __restrict__ Vp,
    float* __restrict__ out) {
  __shared__ __align__(16) int          qfr[NDS * FRAGI];   // 24 KB Q fragments
  __shared__ __align__(16) float        sc[16 * 128];       // 8 KB scores
  __shared__ __align__(16) unsigned int pbuf[16 * 64];      // 4 KB bf16 probs
  __shared__ float alpha[16];
  __shared__ float lsum[16];

  const int tid = threadIdx.x, lane = tid & 31, wave = tid >> 5;
  const int n = lane & 15, hi = lane >> 4;
  const int rt = blockIdx.x;
  const int b  = rt >> 7;                    // 128 row-tiles per batch

  for (int i = tid; i < NDS * FRAGI; i += 256)
    qfr[i] = qfrag[rt * (NDS * FRAGI) + i];
  __syncthreads();

  v8f oacc[6];
#pragma unroll
  for (int t = 0; t < 6; ++t) oacc[t] = vzero8();

  // Distributed online-softmax state: this thread serves row (tid>>4),
  // columns [(tid&15)*8, (tid&15)*8+8) of the current 128-wide k-tile.
  const int srow = tid >> 4;
  const int scol = (tid & 15) * 8;
  float mrun = -3.0e38f, lrun = 0.f;

  for (int kt = 0; kt < 16; ++kt) {
    // ---- scores: this wave's 16 k-columns -------------------------------
    FragF c; c.v = vzero8();
    const int ktrt = b * 128 + kt * 8 + wave;           // K row-tile
    for (int j = 0; j < NDS; ++j) {
      FragBF a, bf;
      const int* ap = qfr + j * FRAGI + lane * 8;
      a.q[0] = *(const int4*)ap;
      a.q[1] = *(const int4*)(ap + 4);
      const int4* kp = (const int4*)(kfrag + ((ktrt * NDS + j) << 8) + lane * 8);
      bf.q[0] = kp[0]; bf.q[1] = kp[1];
      c.v = wmma_bf16(a, bf, c.v);
      if (kt < 15)  // warm L2/WGP$ with next k-tile's fragments (counter-free)
        __builtin_prefetch(kfrag + (((ktrt + 8) * NDS + j) << 8) + lane * 8, 0, 1);
    }
#pragma unroll
    for (int r = 0; r < 8; ++r)
      sc[(r + 8 * hi) * 128 + wave * 16 + n] = c.f[r];
    __syncthreads();  // B1: scores visible; prev-iter P consumers done

    // ---- distributed online softmax -------------------------------------
    {
      float p[8];
#pragma unroll
      for (int i = 0; i < 8; ++i) p[i] = sc[srow * 128 + scol + i];
      float mx = p[0];
#pragma unroll
      for (int i = 1; i < 8; ++i) mx = fmaxf(mx, p[i]);
#pragma unroll
      for (int msk = 8; msk >= 1; msk >>= 1)     // 16-lane max reduction
        mx = fmaxf(mx, __shfl_xor(mx, msk));
      mx = fmaxf(mx, mrun);
      const float al = __expf(mrun - mx);
      float sum = 0.f;
#pragma unroll
      for (int i = 0; i < 8; ++i) { p[i] = __expf(p[i] - mx); sum += p[i]; }
#pragma unroll
      for (int msk = 8; msk >= 1; msk >>= 1)     // 16-lane sum reduction
        sum += __shfl_xor(sum, msk);
      lrun = lrun * al + sum;
      mrun = mx;
#pragma unroll
      for (int i = 0; i < 4; ++i)
        pbuf[srow * 64 + (scol >> 1) + i] = pack2bf(p[2 * i], p[2 * i + 1]);
      if ((tid & 15) == 0) { alpha[srow] = al; lsum[srow] = lrun; }
    }
    __syncthreads();  // B2: P, alpha, lsum visible

    // ---- O update: this wave owns d-cols [wave*96, wave*96+96) ----------
    float a8[8];
#pragma unroll
    for (int r = 0; r < 8; ++r) a8[r] = alpha[r + 8 * hi];
#pragma unroll
    for (int t = 0; t < 6; ++t)
#pragma unroll
      for (int r = 0; r < 8; ++r) oacc[t][r] *= a8[r];

    const int dcol0  = wave * 96 + n;
    const int spbase = (b * SEQ + kt * 128) >> 1;       // global seq-pair base
    for (int p = 0; p < 4; ++p) {
      FragBF pa;
#pragma unroll
      for (int r = 0; r < 8; ++r) {
        const int pr = 8 * hi + r;  // A pair map
        pa.i[r] = (int)pbuf[n * 64 + p * 16 + pr];
      }
#pragma unroll
      for (int t = 0; t < 6; ++t) {
        FragBF vb;
#pragma unroll
        for (int r = 0; r < 8; ++r) {
          const int sp = spbase + p * 16 + 8 * hi + r;  // B pair map: 16*hi+2r over pairs
          vb.i[r] = (int)Vp[(unsigned)sp * DIM + dcol0 + t * 16];
        }
        oacc[t] = wmma_bf16(pa, vb, oacc[t]);
      }
    }
  }

  // ---- epilogue: normalize, residual add, store -------------------------
  float li[8];
#pragma unroll
  for (int r = 0; r < 8; ++r) li[r] = 1.f / lsum[r + 8 * hi];
#pragma unroll
  for (int t = 0; t < 6; ++t) {
    const int col = wave * 96 + t * 16 + n;
#pragma unroll
    for (int r = 0; r < 8; ++r) {
      const int grow = rt * 16 + r + 8 * hi;
      out[grow * DIM + col] = oacc[t][r] * li[r] + x[grow * DIM + col];
    }
  }
}

// ---------------------------------------------------------------------------
// Launch. Workspace layout (dwords):
//   wfrag : 3*48*24*256          =   884,736  ( 3.4 MB)
//   qfrag : 1024*24*256          = 6,291,456  (24.0 MB)
//   kfrag : 6,291,456            (24.0 MB)
//   Vp    : 8192*768             = 6,291,456  (24.0 MB)
// total ~79 MB.
// ---------------------------------------------------------------------------
extern "C" void kernel_launch(void* const* d_in, const int* in_sizes, int n_in,
                              void* d_out, int out_size, void* d_ws, size_t ws_size,
                              hipStream_t stream) {
  const float* x  = (const float*)d_in[0];
  const float* Wq = (const float*)d_in[1];
  const float* bq = (const float*)d_in[2];
  const float* Wk = (const float*)d_in[3];
  const float* bk = (const float*)d_in[4];
  const float* Wv = (const float*)d_in[5];
  const float* bv = (const float*)d_in[6];
  float* out = (float*)d_out;

  int* wfrag = (int*)d_ws;
  int* qfrag = wfrag + 3 * NET * NDS * FRAGI;
  int* kfrag = qfrag + NRT * NDS * FRAGI;
  unsigned int* Vp = (unsigned int*)(kfrag + NRT * NDS * FRAGI);

  wfrag_kernel<<<3 * NET * NDS, 32, 0, stream>>>(Wq, Wk, Wv, wfrag);
  qkv_proj_kernel<<<NRT, 256, 0, stream>>>(x, bq, bk, bv, wfrag, qfrag, kfrag, Vp);
  attn_kernel<<<NRT, 256, 0, stream>>>(x, qfrag, kfrag, Vp, out);
}
